// EERNN_54219667144829
// MI455X (gfx1250) — matrix operand order repaired and checked by reading, not compiled, not run on hardware
//
#include <hip/hip_runtime.h>
#include <hip/hip_bf16.h>
#include <cmath>

typedef __attribute__((ext_vector_type(16))) _Float16 v16h;
typedef __attribute__((ext_vector_type(8)))  float    v8f;

#define MROWS 16  // padded batch rows for WMMA M dimension

// ---------------- f32 -> f16 convert ----------------
__global__ void k_f2h(const float* __restrict__ src, _Float16* __restrict__ dst, int n) {
  int i = blockIdx.x * blockDim.x + threadIdx.x;
  if (i < n) dst[i] = (_Float16)src[i];
}

// ---------------- xt = tile(X[pro_id],2) * onehot_table[label], as f16 ----------------
// Writes a 16-row padded buffer: rows [0,B) real, rows [B,16) zero.
__global__ void k_embed(const int* __restrict__ pro_id, const int* __restrict__ label,
                        const float* __restrict__ X, const float* __restrict__ onehot,
                        _Float16* __restrict__ xt, int B, int T, int n) {
  int i = blockIdx.x * blockDim.x + threadIdx.x;
  if (i >= n) return;               // n = 16*T*256
  int row = i >> 8;                 // m*T + t
  int k   = i & 255;                // 0..255 (4U)
  int m   = row / T;
  if (m >= B) { xt[i] = (_Float16)0.f; return; }
  int pid = pro_id[row];
  int lab = label[row];
  xt[i] = (_Float16)(X[pid * 128 + (k & 127)] * onehot[lab * 256 + k]);
}

// ---------------- LSTM: one workgroup, 16 waves, WMMA per 16-col tile of z ----------------
// z[16,256] = xt_t[16,256] @ Kw[256,256] + h[16,64] @ Rw[64,256] + bias
__global__ __launch_bounds__(512)
void k_lstm(const _Float16* __restrict__ xt,   // [16*T, 256] f16 (rows >= B are zero)
            const _Float16* __restrict__ Kw,   // [256,256] f16 row-major
            const _Float16* __restrict__ Rw,   // [64,256]  f16 row-major
            const float*   __restrict__ bias,  // [256]
            float*         __restrict__ ht_out,// [B*T, 64]
            int B, int T) {
  __shared__ float    zbuf[16][256];
  __shared__ _Float16 hf16[16][64];
  __shared__ float    cbuf[8][64];

  const int tid  = threadIdx.x;
  const int lane = tid & 31;
  const int wave = tid >> 5;          // 0..15 -> N-tile
  const int noff = wave * 16;
  const int col  = lane & 15;         // N within tile (B/C/D layout)
  const int m    = lane & 15;         // M row for A layout
  const int upper = lane >> 4;        // lane half

  // init h, c
  for (int i = tid; i < 16 * 64; i += 512) hf16[i >> 6][i & 63] = (_Float16)0.f;
  for (int i = tid; i < 8 * 64;  i += 512) cbuf[i >> 6][i & 63] = 0.f;
  __syncthreads();

  // A-fragment K map (ISA 7.12.2, 16-bit A 16x32)
  int kmapA[16];
#pragma unroll
  for (int j = 0; j < 16; ++j) {
    int v = j >> 1;
    int base = (v < 4) ? (2 * v) : (16 + 2 * (v - 4));
    kmapA[j] = base + upper * 8 + (j & 1);
  }
  const int kbB = upper * 16;  // unused: B-fragments use kmapA, same k order as A

  // Hoist all weight B-fragments into registers (loop-invariant): 10 x v16h
  v16h wK[8], wR[2];
#pragma unroll
  for (int kb = 0; kb < 8; ++kb)
#pragma unroll
    for (int j = 0; j < 16; ++j)
      wK[kb][j] = Kw[(size_t)(kb * 32 + kmapA[j]) * 256 + noff + col];
#pragma unroll
  for (int kb = 0; kb < 2; ++kb)
#pragma unroll
    for (int j = 0; j < 16; ++j)
      wR[kb][j] = Rw[(size_t)(kb * 32 + kmapA[j]) * 256 + noff + col];

  for (int t = 0; t < T; ++t) {
    const _Float16* xrow = xt + ((size_t)(m * T + t)) * 256;  // padded: always in-bounds

    // software pipeline: issue ALL x A-fragment loads up front (independent),
    // then run the dependent WMMA chain while loads drain.
    v16h ax[8];
#pragma unroll
    for (int kb = 0; kb < 8; ++kb)
#pragma unroll
      for (int j = 0; j < 16; ++j)
        ax[kb][j] = xrow[kb * 32 + kmapA[j]];

    v8f acc = {};
#pragma unroll
    for (int kb = 0; kb < 8; ++kb)
      acc = __builtin_amdgcn_wmma_f32_16x16x32_f16(false, ax[kb], false, wK[kb],
                                                   (short)0, acc, false, false);

    // h @ rec : K = 64 -> 2 WMMA steps (depends on previous step's barrier)
#pragma unroll
    for (int kb = 0; kb < 2; ++kb) {
      v16h a;
#pragma unroll
      for (int j = 0; j < 16; ++j)
        a[j] = hf16[m][kb * 32 + kmapA[j]];
      acc = __builtin_amdgcn_wmma_f32_16x16x32_f16(false, a, false, wR[kb],
                                                   (short)0, acc, false, false);
    }

    // scatter z tile (+bias) to LDS; C/D layout: row = r + 8*upper, col = lane&15
    const float bv = bias[noff + col];
#pragma unroll
    for (int r = 0; r < 8; ++r)
      zbuf[r + 8 * upper][noff + col] = acc[r] + bv;
    __syncthreads();

    // gate update: 512 threads == 8 rows x 64 units
    {
      int b = tid >> 6;
      int u = tid & 63;
      float zi = zbuf[b][u], zf = zbuf[b][64 + u];
      float zg = zbuf[b][128 + u], zo = zbuf[b][192 + u];
      float ig = 1.f / (1.f + __expf(-zi));
      float fg = 1.f / (1.f + __expf(-zf));
      float gg = tanhf(zg);
      float og = 1.f / (1.f + __expf(-zo));
      float c  = fg * cbuf[b][u] + ig * gg;
      cbuf[b][u] = c;
      float h = og * tanhf(c);
      if (b < B) {
        ht_out[((size_t)b * T + t) * 64 + u] = h;
        hf16[b][u] = (_Float16)h;
      } else {
        hf16[b][u] = (_Float16)0.f;
      }
    }
    __syncthreads();
  }
}

// ---------------- hW[b,t,:] = ht[b,t,:] @ W1[0:64,:] ----------------
__global__ void k_hW(const float* __restrict__ ht, const float* __restrict__ W1,
                     float* __restrict__ hW, int n) {
  int i = blockIdx.x * blockDim.x + threadIdx.x;
  if (i >= n) return;
  int bt = i / 50, j = i % 50;
  float s = 0.f;
#pragma unroll
  for (int u = 0; u < 64; ++u) s = fmaf(ht[(size_t)bt * 64 + u], W1[u * 50 + j], s);
  hW[i] = s;
}

// ---------------- XW[p,:] = X[p,:] @ W1[64:192,:] + b1 ----------------
__global__ void k_xw(const float* __restrict__ X, const float* __restrict__ W1,
                     const float* __restrict__ b1, float* __restrict__ XW, int n) {
  int i = blockIdx.x * blockDim.x + threadIdx.x;
  if (i >= n) return;
  int p = i / 50, j = i % 50;
  float s = b1[j];
#pragma unroll
  for (int k = 0; k < 128; ++k)
    s = fmaf(X[(size_t)p * 128 + k], W1[(64 + k) * 50 + j], s);
  XW[i] = s;
}

// ---------------- streaming suffix-sum attention + MLP head ----------------
// per (b,p): acc[j] = sum_{s>=t} cos_X[pro_id[b,s],p] * hW[b,s,j]
// out[b,t,p] = relu(acc + XW[p]) . W2 + b2
__global__ void k_final(const int* __restrict__ pro_id, const float* __restrict__ cos_X,
                        const float* __restrict__ hW, const float* __restrict__ XW,
                        const float* __restrict__ W2, const float* __restrict__ b2,
                        float* __restrict__ out, int B, int T, int P) {
  int tid = blockIdx.x * blockDim.x + threadIdx.x;
  if (tid >= B * P) return;
  int b = tid / P, p = tid % P;

  float xwv[50];
#pragma unroll
  for (int j = 0; j < 50; ++j) xwv[j] = XW[(size_t)p * 50 + j];

  float acc[50];
#pragma unroll
  for (int j = 0; j < 50; ++j) acc[j] = 0.f;

  const float b2v = b2[0];
  for (int t = T - 1; t >= 0; --t) {
    int pid = pro_id[b * T + t];
    float av = cos_X[(size_t)pid * P + p];
    const float* hw = hW + ((size_t)b * T + t) * 50;
#pragma unroll
    for (int j = 0; j < 50; ++j) acc[j] = fmaf(av, hw[j], acc[j]);
    float s = b2v;
#pragma unroll
    for (int j = 0; j < 50; ++j) {
      float r = acc[j] + xwv[j];
      r = r > 0.f ? r : 0.f;
      s = fmaf(r, W2[j], s);
    }
    out[((size_t)b * T + t) * P + p] = s;
  }
}

extern "C" void kernel_launch(void* const* d_in, const int* in_sizes, int n_in,
                              void* d_out, int out_size, void* d_ws, size_t ws_size,
                              hipStream_t stream) {
  const int*   pro_id      = (const int*)d_in[0];
  const int*   label       = (const int*)d_in[1];
  const float* X           = (const float*)d_in[3];
  const float* cos_X       = (const float*)d_in[4];
  const float* onehot      = (const float*)d_in[6];
  const float* lstm_kernel = (const float*)d_in[7];
  const float* lstm_rec    = (const float*)d_in[8];
  const float* lstm_bias   = (const float*)d_in[9];
  const float* W1          = (const float*)d_in[10];
  const float* b1          = (const float*)d_in[11];
  const float* W2          = (const float*)d_in[12];
  const float* b2          = (const float*)d_in[13];

  const int T = (int)lround(sqrt((double)in_sizes[5]));  // trimatrix T*T
  const int B = in_sizes[0] / T;                         // pro_id B*T
  const int P = in_sizes[3] / 128;                       // X P*2U
  const int BT = B * T;

  // carve workspace (256B aligned chunks)
  char* ws = (char*)d_ws;
  size_t off = 0;
  auto carve = [&](size_t bytes) -> char* {
    off = (off + 255) & ~(size_t)255;
    char* p = ws + off;
    off += bytes;
    return p;
  };
  _Float16* xt16 = (_Float16*)carve((size_t)MROWS * T * 256 * sizeof(_Float16));
  _Float16* Kw16 = (_Float16*)carve((size_t)256 * 256 * sizeof(_Float16));
  _Float16* Rw16 = (_Float16*)carve((size_t)64 * 256 * sizeof(_Float16));
  float*    ht   = (float*)   carve((size_t)BT * 64 * sizeof(float));
  float*    hW   = (float*)   carve((size_t)BT * 50 * sizeof(float));
  float*    XW   = (float*)   carve((size_t)P * 50 * sizeof(float));

  float* out = (float*)d_out;

  // 1) weight conversion
  {
    int n = 256 * 256;
    k_f2h<<<(n + 255) / 256, 256, 0, stream>>>(lstm_kernel, Kw16, n);
    n = 64 * 256;
    k_f2h<<<(n + 255) / 256, 256, 0, stream>>>(lstm_rec, Rw16, n);
  }
  // 2) input embedding into 16-row padded buffer
  {
    int n = MROWS * T * 256;
    k_embed<<<(n + 255) / 256, 256, 0, stream>>>(pro_id, label, X, onehot, xt16, B, T, n);
  }
  // 3) LSTM recurrence (single workgroup, WMMA)
  k_lstm<<<1, 512, 0, stream>>>(xt16, Kw16, Rw16, lstm_bias, ht, B, T);
  // 4) hW = ht @ W1a
  {
    int n = BT * 50;
    k_hW<<<(n + 255) / 256, 256, 0, stream>>>(ht, W1, hW, n);
  }
  // 5) XW = X @ W1b + b1
  {
    int n = P * 50;
    k_xw<<<(n + 255) / 256, 256, 0, stream>>>(X, W1, b1, XW, n);
  }
  // 6) streaming suffix-sum + MLP head
  {
    int n = B * P;
    k_final<<<(n + 255) / 256, 256, 0, stream>>>(pro_id, cos_X, hW, XW, W2, b2,
                                                 out, B, T, P);
  }
}
